// MPNNLayer_23691039604762
// MI455X (gfx1250) — matrix operand order repaired and checked by reading, not compiled, not run on hardware
//
#include <hip/hip_runtime.h>

// ---------------- CDNA5 WMMA types ----------------
typedef __attribute__((ext_vector_type(16))) __bf16 v16bf;
typedef __attribute__((ext_vector_type(8)))  __bf16 v8bf;
typedef __attribute__((ext_vector_type(8)))  float  v8f;

#define BATCH  2
#define NNODE  2048
#define KNBR   48
#define CDIM   128

#define XSTRIDE 392   // halfs per row of 48x384 staging (pad 8)
#define HSTRIDE 136   // halfs per row of 48x128 staging (pad 8)
#define OSTRIDE 132   // floats per row of 48x128 pre-LN output (pad 4)

#define NODE_SMEM 52480
#define EDGE_SMEM 76224

static __device__ __forceinline__ unsigned short f2bf(float f) {
  unsigned u = __builtin_bit_cast(unsigned, f);
  unsigned r = u + 0x7FFFu + ((u >> 16) & 1u);   // round-to-nearest-even
  return (unsigned short)(r >> 16);
}

static __device__ __forceinline__ float gelu_tanh(float x) {
  float x3 = x * x * x;
  return 0.5f * x * (1.0f + tanhf(0.7978845608028654f * (x + 0.044715f * x3)));
}

// A-fragment (16x32 bf16, MxK) per the CDNA5 layout:
// lanes 0-15 : row M=lane,    K in {0..7, 16..23}
// lanes 16-31: row M=lane-16, K in {8..15, 24..31}
static __device__ __forceinline__ v16bf load_a(const unsigned short* p) {
  v8bf lo = *(const v8bf*)p;          // 8 halfs at kbase
  v8bf hi = *(const v8bf*)(p + 16);   // 8 halfs at kbase+16
  return __builtin_shufflevector(lo, hi, 0, 1, 2, 3, 4, 5, 6, 7,
                                         8, 9, 10, 11, 12, 13, 14, 15);
}

static __device__ __forceinline__ void init_acc(v8f acc[3], float b) {
#pragma unroll
  for (int m = 0; m < 3; ++m)
#pragma unroll
    for (int i = 0; i < 8; ++i) acc[m][i] = b;
}

// 48 x (32*ktiles) [A in LDS] x (..x16) [packed W tile for this wave] -> acc[3]
// All three A fragments are loaded into distinct registers BEFORE the WMMAs so
// the 6 ds_load_b128 can be in flight together and retire under staged waits.
static __device__ __forceinline__ void gemm48(const unsigned short* __restrict__ A,
                                              int astride, int ktiles,
                                              const v16bf* __restrict__ Wp,
                                              int wave, int lane, v8f acc[3]) {
  const int lrow = lane & 15;
  const int khi  = (lane & 16) ? 8 : 0;
  for (int kt = 0; kt < ktiles; ++kt) {
    v16bf bf = Wp[((kt << 3) + wave) * 32 + lane];          // 32B contiguous/lane
    const unsigned short* ab = A + (kt << 5) + khi;
    v16bf a0 = load_a(ab + (lrow) * astride);
    v16bf a1 = load_a(ab + (16 + lrow) * astride);
    v16bf a2 = load_a(ab + (32 + lrow) * astride);
    acc[0] = __builtin_amdgcn_wmma_f32_16x16x32_bf16(
        false, a0, false, bf, (short)0, acc[0], false, false);
    acc[1] = __builtin_amdgcn_wmma_f32_16x16x32_bf16(
        false, a1, false, bf, (short)0, acc[1], false, false);
    acc[2] = __builtin_amdgcn_wmma_f32_16x16x32_bf16(
        false, a2, false, bf, (short)0, acc[2], false, false);
  }
}

static __device__ __forceinline__ void store_gelu(unsigned short* __restrict__ Hs,
                                                  int ncol, int hi8, const v8f acc[3]) {
#pragma unroll
  for (int m = 0; m < 3; ++m)
#pragma unroll
    for (int r = 0; r < 8; ++r)
      Hs[(m * 16 + hi8 + r) * HSTRIDE + ncol] = f2bf(gelu_tanh(acc[m][r]));
}

// ---------------- weight packing: f32 [K,128] -> bf16 B-fragment tiles ----------------
// Packed layout: ((kt*8 + nt)*32 + lane)*16 + e, where lane holds column
// n = nt*16 + lane%16 and k = kt*32 + (lane<16 ? 0 : 8) + (e&7) + (e<8 ? 0 : 16).
__global__ void __launch_bounds__(256)
pack_w_kernel(const float* __restrict__ W, unsigned short* __restrict__ dst, int kdim) {
  int idx = blockIdx.x * 256 + threadIdx.x;
  int total = (kdim >> 5) * 8 * 32 * 16;
  if (idx >= total) return;
  int e    = idx & 15;
  int lane = (idx >> 4) & 31;
  int tile = idx >> 9;
  int nt = tile & 7;
  int kt = tile >> 3;
  int n = nt * 16 + (lane & 15);
  int k = kt * 32 + ((lane & 16) ? 8 : 0) + (e & 7) + ((e & 8) ? 16 : 0);
  dst[idx] = f2bf(W[k * CDIM + n]);
}

// ---------------- node update: message MLP + sum/60 + LN1 + FFN + LN2 ----------------
__global__ void __launch_bounds__(256)
node_kernel(const float* __restrict__ hV, const float* __restrict__ hE,
            const int* __restrict__ topo,
            const unsigned short* __restrict__ W1p, const unsigned short* __restrict__ W2p,
            const unsigned short* __restrict__ W3p,
            const float* __restrict__ b1, const float* __restrict__ b2,
            const float* __restrict__ b3,
            const float* __restrict__ fW1, const float* __restrict__ fb1,
            const float* __restrict__ fW2, const float* __restrict__ fb2,
            const float* __restrict__ ln1s, const float* __restrict__ ln1b,
            const float* __restrict__ ln2s, const float* __restrict__ ln2b,
            float* __restrict__ outV, unsigned short* __restrict__ hVp) {
  extern __shared__ __align__(16) char smem[];
  float* redT = (float*)smem;         // 128
  float* redA = redT + 128;           // 128
  float* redB = redA + 128;           // 128
  float* redS = redB + 128;           // 8
  float* redQ = redS + 8;             // 8
  int*   topoS = (int*)(redQ + 8);    // 48
  unsigned short* Xs = (unsigned short*)(smem + 1792);
  unsigned short* Hs = (unsigned short*)(smem + 1792 + KNBR * XSTRIDE * 2);

  const int node = blockIdx.x;
  const int b = node / NNODE;
  const int t = threadIdx.x;
  const int wave = t >> 5, lane = t & 31;
  const int lrow = lane & 15;
  const int hi8  = (lane & 16) ? 8 : 0;
  const int ncol = (wave << 4) | lrow;

  if (t < KNBR) topoS[t] = b * NNODE + topo[node * KNBR + t];
  __syncthreads();

  // build X = [hV_i | hV_j | hE_ij] as bf16 in LDS (48 x 384)
  for (int idx = t; idx < KNBR * 96; idx += 256) {
    int r = idx / 96, ch = idx % 96;
    float4 v;
    if (ch < 32)      v = ((const float4*)(hV + (size_t)node * CDIM))[ch];
    else if (ch < 64) v = ((const float4*)(hV + (size_t)topoS[r] * CDIM))[ch - 32];
    else              v = ((const float4*)(hE + ((size_t)node * KNBR + r) * CDIM))[ch - 64];
    union { unsigned short h[4]; uint2 u; } pk;
    pk.h[0] = f2bf(v.x); pk.h[1] = f2bf(v.y); pk.h[2] = f2bf(v.z); pk.h[3] = f2bf(v.w);
    *(uint2*)(Xs + r * XSTRIDE + ch * 4) = pk.u;
  }
  __syncthreads();

  v8f acc[3];
  // layer 1: 48x384 @ 384x128
  init_acc(acc, b1[ncol]);
  gemm48(Xs, XSTRIDE, 12, (const v16bf*)W1p, wave, lane, acc);
  store_gelu(Hs, ncol, hi8, acc);
  __syncthreads();

  // layer 2: 48x128 @ 128x128
  init_acc(acc, b2[ncol]);
  gemm48(Hs, HSTRIDE, 4, (const v16bf*)W2p, wave, lane, acc);
  __syncthreads();                 // all waves done reading H1
  store_gelu(Hs, ncol, hi8, acc);
  __syncthreads();

  // layer 3: 48x128 @ 128x128 (bias folded in -> summed gives 48*b3)
  init_acc(acc, b3[ncol]);
  gemm48(Hs, HSTRIDE, 4, (const v16bf*)W3p, wave, lane, acc);

  // sum messages over 48 neighbors, / 60, + residual
  float partial = 0.0f;
#pragma unroll
  for (int m = 0; m < 3; ++m)
#pragma unroll
    for (int r = 0; r < 8; ++r) partial += acc[m][r];
  partial += __shfl_xor(partial, 16);
  float dh = partial * (1.0f / 60.0f);
  if (lane < 16) redT[ncol] = hV[(size_t)node * CDIM + ncol] + dh;
  __syncthreads();

  // LayerNorm 1
  float x = (t < 128) ? redT[t] : 0.0f;
  float s = x, q = x * x;
#pragma unroll
  for (int off = 16; off; off >>= 1) { s += __shfl_xor(s, off); q += __shfl_xor(q, off); }
  if (lane == 0) { redS[wave] = s; redQ[wave] = q; }
  __syncthreads();
  float mu = (redS[0] + redS[1] + redS[2] + redS[3]) * (1.0f / 128.0f);
  float qq = (redQ[0] + redQ[1] + redQ[2] + redQ[3]) * (1.0f / 128.0f);
  float rstd = rsqrtf(qq - mu * mu + 1e-6f);
  float hv1 = 0.0f;
  if (t < 128) {
    hv1 = (x - mu) * rstd * ln1s[t] + ln1b[t];
    redA[t] = hv1;
  }
  __syncthreads();

  // position-wise FFN (tiny: 2 * 128x128 GEMV) in f32
  if (t < 128) {
    float a = fb1[t];
    for (int k2 = 0; k2 < 128; ++k2) a = fmaf(redA[k2], fW1[k2 * CDIM + t], a);
    redB[t] = gelu_tanh(a);
  }
  __syncthreads();
  float y = 0.0f;
  if (t < 128) {
    float a = fb2[t];
    for (int k2 = 0; k2 < 128; ++k2) a = fmaf(redB[k2], fW2[k2 * CDIM + t], a);
    y = hv1 + a;
  }

  // LayerNorm 2 -> final hV (f32 out + bf16 copy for edge pass)
  s = y; q = y * y;
#pragma unroll
  for (int off = 16; off; off >>= 1) { s += __shfl_xor(s, off); q += __shfl_xor(q, off); }
  if (lane == 0) { redS[wave] = s; redQ[wave] = q; }
  __syncthreads();
  mu = (redS[0] + redS[1] + redS[2] + redS[3]) * (1.0f / 128.0f);
  qq = (redQ[0] + redQ[1] + redQ[2] + redQ[3]) * (1.0f / 128.0f);
  rstd = rsqrtf(qq - mu * mu + 1e-6f);
  if (t < 128) {
    float o = (y - mu) * rstd * ln2s[t] + ln2b[t];
    outV[(size_t)node * CDIM + t] = o;
    hVp[(size_t)node * CDIM + t] = f2bf(o);
  }
}

// ---------------- edge update: edge MLP + residual + LN3 ----------------
__global__ void __launch_bounds__(256)
edge_kernel(const float* __restrict__ hE, const int* __restrict__ topo,
            const unsigned short* __restrict__ hVp,
            const unsigned short* __restrict__ W1p, const unsigned short* __restrict__ W2p,
            const unsigned short* __restrict__ W3p,
            const float* __restrict__ b1, const float* __restrict__ b2,
            const float* __restrict__ b3,
            const float* __restrict__ ln3s, const float* __restrict__ ln3b,
            float* __restrict__ outE) {
  extern __shared__ __align__(16) char smem[];
  float* OUTs = (float*)smem;                                   // 48 x 132 f32
  int*   topoS = (int*)(smem + KNBR * OSTRIDE * 4);             // 48
  unsigned short* Xs = (unsigned short*)(smem + 25536);
  unsigned short* Hs = (unsigned short*)(smem + 25536 + KNBR * XSTRIDE * 2);

  const int node = blockIdx.x;
  const int b = node / NNODE;
  const int t = threadIdx.x;
  const int wave = t >> 5, lane = t & 31;
  const int lrow = lane & 15;
  const int hi8  = (lane & 16) ? 8 : 0;
  const int ncol = (wave << 4) | lrow;

  if (t < KNBR) topoS[t] = b * NNODE + topo[node * KNBR + t];
  __syncthreads();

  // X cols 0..255 from bf16 hV (new), 16B chunks
  for (int idx = t; idx < KNBR * 32; idx += 256) {
    int r = idx >> 5, ch = idx & 31;
    int c = ch << 3;
    const unsigned short* src = (c < 128)
        ? (hVp + (size_t)node * CDIM + c)
        : (hVp + (size_t)topoS[r] * CDIM + (c - 128));
    *(uint4*)(Xs + r * XSTRIDE + c) = *(const uint4*)src;
  }
  // X cols 256..383 from f32 h_E
  for (int idx = t; idx < KNBR * 32; idx += 256) {
    int r = idx >> 5, ch = idx & 31;
    float4 v = ((const float4*)(hE + ((size_t)node * KNBR + r) * CDIM))[ch];
    union { unsigned short h[4]; uint2 u; } pk;
    pk.h[0] = f2bf(v.x); pk.h[1] = f2bf(v.y); pk.h[2] = f2bf(v.z); pk.h[3] = f2bf(v.w);
    *(uint2*)(Xs + r * XSTRIDE + 256 + (ch << 2)) = pk.u;
  }
  __syncthreads();

  v8f acc[3];
  init_acc(acc, b1[ncol]);
  gemm48(Xs, XSTRIDE, 12, (const v16bf*)W1p, wave, lane, acc);
  store_gelu(Hs, ncol, hi8, acc);
  __syncthreads();

  init_acc(acc, b2[ncol]);
  gemm48(Hs, HSTRIDE, 4, (const v16bf*)W2p, wave, lane, acc);
  __syncthreads();
  store_gelu(Hs, ncol, hi8, acc);
  __syncthreads();

  init_acc(acc, b3[ncol]);
  gemm48(Hs, HSTRIDE, 4, (const v16bf*)W3p, wave, lane, acc);

  // residual add (f32) into LDS for row-wise LayerNorm
#pragma unroll
  for (int m = 0; m < 3; ++m)
#pragma unroll
    for (int r = 0; r < 8; ++r) {
      int row = m * 16 + hi8 + r;
      OUTs[row * OSTRIDE + ncol] =
          acc[m][r] + hE[((size_t)node * KNBR + row) * CDIM + ncol];
    }
  __syncthreads();

  // per-edge-row LayerNorm over 128, 4 lanes per row
  int row = t >> 2, sub = t & 3;
  if (row < KNBR) {
    float s = 0.0f, q = 0.0f;
    const float* rp = OUTs + row * OSTRIDE;
    for (int j = sub * 32; j < sub * 32 + 32; ++j) { float v = rp[j]; s += v; q += v * v; }
    s += __shfl_xor(s, 1); q += __shfl_xor(q, 1);
    s += __shfl_xor(s, 2); q += __shfl_xor(q, 2);
    float mu = s * (1.0f / 128.0f);
    float rstd = rsqrtf(q * (1.0f / 128.0f) - mu * mu + 1e-6f);
    float* op = outE + ((size_t)node * KNBR + row) * CDIM;
    for (int j = sub * 32; j < sub * 32 + 32; ++j)
      op[j] = (rp[j] - mu) * rstd * ln3s[j] + ln3b[j];
  }
}

// ---------------- launch ----------------
extern "C" void kernel_launch(void* const* d_in, const int* in_sizes, int n_in,
                              void* d_out, int out_size, void* d_ws, size_t ws_size,
                              hipStream_t stream) {
  (void)in_sizes; (void)n_in; (void)out_size; (void)ws_size;
  const float* hV   = (const float*)d_in[0];
  const float* hE   = (const float*)d_in[1];
  const int*   topo = (const int*)d_in[2];
  const float* nW1 = (const float*)d_in[3];  const float* nb1 = (const float*)d_in[4];
  const float* nW2 = (const float*)d_in[5];  const float* nb2 = (const float*)d_in[6];
  const float* nW3 = (const float*)d_in[7];  const float* nb3 = (const float*)d_in[8];
  const float* eW1 = (const float*)d_in[9];  const float* eb1 = (const float*)d_in[10];
  const float* eW2 = (const float*)d_in[11]; const float* eb2 = (const float*)d_in[12];
  const float* eW3 = (const float*)d_in[13]; const float* eb3 = (const float*)d_in[14];
  const float* fW1 = (const float*)d_in[15]; const float* fb1 = (const float*)d_in[16];
  const float* fW2 = (const float*)d_in[17]; const float* fb2 = (const float*)d_in[18];
  const float* ln1s = (const float*)d_in[19]; const float* ln1b = (const float*)d_in[20];
  const float* ln2s = (const float*)d_in[21]; const float* ln2b = (const float*)d_in[22];
  const float* ln3s = (const float*)d_in[23]; const float* ln3b = (const float*)d_in[24];

  float* outV = (float*)d_out;
  float* outE = outV + (size_t)BATCH * NNODE * CDIM;

  char* ws = (char*)d_ws;
  unsigned short* hVp  = (unsigned short*)ws;                       // 1,048,576 B
  unsigned short* nW1p = (unsigned short*)(ws + 1048576);           //    98,304 B
  unsigned short* nW2p = (unsigned short*)(ws + 1048576 + 98304);   //    32,768 B
  unsigned short* nW3p = (unsigned short*)(ws + 1048576 + 131072);  //    32,768 B
  unsigned short* eW1p = (unsigned short*)(ws + 1048576 + 163840);  //    98,304 B
  unsigned short* eW2p = (unsigned short*)(ws + 1048576 + 262144);  //    32,768 B
  unsigned short* eW3p = (unsigned short*)(ws + 1048576 + 294912);  //    32,768 B

  pack_w_kernel<<<192, 256, 0, stream>>>(nW1, nW1p, 384);
  pack_w_kernel<<<64,  256, 0, stream>>>(nW2, nW2p, 128);
  pack_w_kernel<<<64,  256, 0, stream>>>(nW3, nW3p, 128);
  pack_w_kernel<<<192, 256, 0, stream>>>(eW1, eW1p, 384);
  pack_w_kernel<<<64,  256, 0, stream>>>(eW2, eW2p, 128);
  pack_w_kernel<<<64,  256, 0, stream>>>(eW3, eW3p, 128);

  const int nblocks = BATCH * NNODE;
  node_kernel<<<nblocks, 256, NODE_SMEM, stream>>>(
      hV, hE, topo, nW1p, nW2p, nW3p, nb1, nb2, nb3,
      fW1, fb1, fW2, fb2, ln1s, ln1b, ln2s, ln2b, outV, hVp);
  edge_kernel<<<nblocks, 256, EDGE_SMEM, stream>>>(
      hE, topo, hVp, eW1p, eW2p, eW3p, eb1, eb2, eb3, ln3s, ln3b, outE);
}
